// MultiheadAttention_79242146611891
// MI455X (gfx1250) — matrix operand order repaired and checked by reading, not compiled, not run on hardware
//
#include <hip/hip_runtime.h>
#include <hip/hip_bf16.h>

#define S_TOT 1024
#define BSZ   16
#define EMB   1024
#define HEADS 16
#define HD    64

typedef __attribute__((ext_vector_type(16))) __bf16 v16bf;
typedef __attribute__((ext_vector_type(8)))  float  v8f;

__device__ __forceinline__ __bf16 f2bf(float f) {
  unsigned u = __builtin_bit_cast(unsigned, f);
  u += 0x7FFFu + ((u >> 16) & 1u);               // round-to-nearest-even
  unsigned short h = (unsigned short)(u >> 16);
  return __builtin_bit_cast(__bf16, h);
}

__device__ __forceinline__ v8f wmma_bf16(v16bf a, v16bf b, v8f c) {
  return __builtin_amdgcn_wmma_f32_16x16x32_bf16(false, a, false, b, (short)0, c, false, false);
}

// A-matrix (16x32 bf16) K index for half h (0..15) in this lane (contiguous order):
// lanes 0-15: K 0..15 ; lanes 16-31: K 16..31
__device__ __forceinline__ int a_kidx(int h, int lane) {
  return ((lane >> 4) << 4) | h;
}

// ---------------------------------------------------------------------------
// K1: fused in-projection. X (tokens x E) @ W^T (E x 3E) + b, bf16 WMMA.
// Writes Q (scaled 1/8), K, V head-major: [bh=b*16+h][s][64] in bf16.
// ---------------------------------------------------------------------------
__global__ __launch_bounds__(256) void proj_kernel(
    const float* __restrict__ xq, const float* __restrict__ xk,
    const float* __restrict__ xv, const float* __restrict__ w,
    const float* __restrict__ bias,
    __bf16* __restrict__ Qb, __bf16* __restrict__ Kb, __bf16* __restrict__ Vb,
    int tok_base) {
  __shared__ __align__(32) __bf16 Alds[128 * 32];
  __shared__ __align__(32) __bf16 Blds[64 * 32];
  const int tid = threadIdx.x, lane = tid & 31, wv = tid >> 5;
  const int m0 = blockIdx.x * 128;     // token tile within segment
  const int n0 = blockIdx.y * 64;      // within 3E
  const int sec = n0 >> 10;            // 0=q,1=k,2=v (uniform per block)
  const float* __restrict__ X = (sec == 0) ? xq : (sec == 1) ? xk : xv;
  X += (size_t)tok_base * EMB;
  __bf16* __restrict__ Dst = (sec == 0) ? Qb : (sec == 1) ? Kb : Vb;
  const float scale = (sec == 0) ? 0.125f : 1.0f;   // 1/sqrt(64)

  v8f acc[4] = {};
  for (int k0 = 0; k0 < EMB; k0 += 32) {
    for (int idx = tid; idx < 128 * 32; idx += 256) {
      int i = idx >> 5, j = idx & 31;
      Alds[idx] = f2bf(X[(size_t)(m0 + i) * EMB + k0 + j]);
    }
    for (int idx = tid; idx < 64 * 32; idx += 256) {
      int n = idx >> 5, k = idx & 31;
      Blds[idx] = f2bf(w[(size_t)(n0 + n) * EMB + k0 + k]);
    }
    // prefetch next K-step tiles toward the WGP while we compute
    int kn = (k0 + 32 < EMB) ? k0 + 32 : k0;
    __builtin_prefetch(&X[(size_t)(m0 + (tid >> 1)) * EMB + kn + (tid & 1) * 16], 0, 3);
    __builtin_prefetch(&w[(size_t)(n0 + (tid >> 2)) * EMB + kn + (tid & 3) * 8], 0, 3);
    __syncthreads();
    v16bf a;
#pragma unroll
    for (int h = 0; h < 16; ++h)
      a[h] = Alds[(wv * 16 + (lane & 15)) * 32 + a_kidx(h, lane)];
    v16bf bfr[4];
#pragma unroll
    for (int nt = 0; nt < 4; ++nt)
      bfr[nt] = *reinterpret_cast<const v16bf*>(
          &Blds[(nt * 16 + (lane & 15)) * 32 + ((lane >> 4) << 4)]);
#pragma unroll
    for (int nt = 0; nt < 4; ++nt)
      acc[nt] = wmma_bf16(a, bfr[nt], acc[nt]);
    __syncthreads();
  }
#pragma unroll
  for (int nt = 0; nt < 4; ++nt)
#pragma unroll
    for (int r = 0; r < 8; ++r) {
      int m = m0 + wv * 16 + r + ((lane >> 4) << 3);
      int n = n0 + nt * 16 + (lane & 15);
      float val = (acc[nt][r] + bias[n]) * scale;
      int e = n & 1023, head = e >> 6, d = e & 63;
      int t = tok_base + m;
      int b = t & 15;        // batch
      int s = t >> 4;        // global seq pos
      Dst[((size_t)(b * HEADS + head) * S_TOT + s) * HD + d] = f2bf(val);
    }
}

// ---------------------------------------------------------------------------
// K2: softmax row stats (max, sum-exp) per (bh,row), scores via WMMA.
// ---------------------------------------------------------------------------
__global__ __launch_bounds__(256) void stats_kernel(
    const __bf16* __restrict__ Qb, const __bf16* __restrict__ Kb,
    const float* __restrict__ mask, float* __restrict__ stats) {
  const int tid = threadIdx.x, lane = tid & 31, wv = tid >> 5;
  const int bh = blockIdx.y;
  const int q0 = blockIdx.x * 128 + wv * 16;
  const __bf16* Qrow = Qb + (size_t)bh * S_TOT * HD;
  const __bf16* Krow = Kb + (size_t)bh * S_TOT * HD;

  v16bf a0, a1;
  int qr = q0 + (lane & 15);
#pragma unroll
  for (int h = 0; h < 16; ++h) {
    int k = a_kidx(h, lane);
    a0[h] = Qrow[(size_t)qr * HD + k];
    a1[h] = Qrow[(size_t)qr * HD + 32 + k];
  }
  float mrun[8], lrun[8];
#pragma unroll
  for (int r = 0; r < 8; ++r) { mrun[r] = -1e30f; lrun[r] = 0.f; }

  for (int n0 = 0; n0 < S_TOT; n0 += 16) {
    int key = n0 + (lane & 15);
    const __bf16* kp = &Krow[(size_t)key * HD + ((lane >> 4) << 4)];
    v16bf b0 = *reinterpret_cast<const v16bf*>(kp);
    v16bf b1 = *reinterpret_cast<const v16bf*>(kp + 32);
    v8f z = {};
    v8f sc = wmma_bf16(a0, b0, z);
    sc = wmma_bf16(a1, b1, sc);
#pragma unroll
    for (int r = 0; r < 8; ++r) {
      int row = q0 + r + ((lane >> 4) << 3);
      float val = sc[r] + mask[(size_t)row * S_TOT + key];
      float nm = fmaxf(mrun[r], val);
      lrun[r] = lrun[r] * __expf(mrun[r] - nm) + __expf(val - nm);
      mrun[r] = nm;
    }
  }
#pragma unroll
  for (int r = 0; r < 8; ++r) {
    float m = mrun[r], l = lrun[r];
#pragma unroll
    for (int off = 1; off < 16; off <<= 1) {
      float om = __shfl_xor(m, off, 32);
      float ol = __shfl_xor(l, off, 32);
      float nm = fmaxf(m, om);
      l = l * __expf(m - nm) + ol * __expf(om - nm);
      m = nm;
    }
    if ((lane & 15) == 0) {
      int row = q0 + r + ((lane >> 4) << 3);
      stats[((size_t)bh * S_TOT + row) * 2 + 0] = m;
      stats[((size_t)bh * S_TOT + row) * 2 + 1] = l;
    }
  }
}

// ---------------------------------------------------------------------------
// K3: ctx = softmax(QK^T + mask) @ V per (bh, 128 query rows).
// Scores via WMMA, P re-laid-out through LDS, P@V via WMMA.
// ctx -> bf16 token-major [t = s*B + b][E].
// ---------------------------------------------------------------------------
__global__ __launch_bounds__(256) void attn_kernel(
    const __bf16* __restrict__ Qb, const __bf16* __restrict__ Kb,
    const __bf16* __restrict__ Vb, const float* __restrict__ mask,
    const float* __restrict__ stats, __bf16* __restrict__ ctxb) {
  __shared__ __align__(32) __bf16 Plds[8 * 16 * 32];
  const int tid = threadIdx.x, lane = tid & 31, wv = tid >> 5;
  const int bh = blockIdx.y;
  const int bidx = bh >> 4, head = bh & 15;
  const int q0 = blockIdx.x * 128 + wv * 16;
  const __bf16* Qrow = Qb + (size_t)bh * S_TOT * HD;
  const __bf16* Krow = Kb + (size_t)bh * S_TOT * HD;
  const __bf16* Vrow = Vb + (size_t)bh * S_TOT * HD;

  v16bf a0, a1;
  int qr = q0 + (lane & 15);
#pragma unroll
  for (int h = 0; h < 16; ++h) {
    int k = a_kidx(h, lane);
    a0[h] = Qrow[(size_t)qr * HD + k];
    a1[h] = Qrow[(size_t)qr * HD + 32 + k];
  }
  float M[8], IL[8];
#pragma unroll
  for (int r = 0; r < 8; ++r) {
    int row = q0 + r + ((lane >> 4) << 3);
    M[r] = stats[((size_t)bh * S_TOT + row) * 2 + 0];
    float l = stats[((size_t)bh * S_TOT + row) * 2 + 1];
    IL[r] = (l > 0.f) ? (1.f / l) : 0.f;     // nan_to_num guard
  }

  v8f cacc[4] = {};
  __bf16* Pw = &Plds[wv * 512];

  for (int c0 = 0; c0 < S_TOT; c0 += 32) {
    __syncthreads();                          // prior P reads done
#pragma unroll
    for (int sub = 0; sub < 2; ++sub) {
      int n0 = c0 + sub * 16;
      int key = n0 + (lane & 15);
      const __bf16* kp = &Krow[(size_t)key * HD + ((lane >> 4) << 4)];
      v16bf b0 = *reinterpret_cast<const v16bf*>(kp);
      v16bf b1 = *reinterpret_cast<const v16bf*>(kp + 32);
      v8f z = {};
      v8f sc = wmma_bf16(a0, b0, z);
      sc = wmma_bf16(a1, b1, sc);
#pragma unroll
      for (int r = 0; r < 8; ++r) {
        int rloc = r + ((lane >> 4) << 3);
        int row = q0 + rloc;
        float val = sc[r] + mask[(size_t)row * S_TOT + key];
        float p = __expf(val - M[r]) * IL[r];
        Pw[rloc * 32 + sub * 16 + (lane & 15)] = f2bf(p);
      }
    }
    __syncthreads();                          // P visible
    v16bf pa;
#pragma unroll
    for (int h = 0; h < 16; ++h)
      pa[h] = Pw[(lane & 15) * 32 + a_kidx(h, lane)];
#pragma unroll
    for (int nt = 0; nt < 4; ++nt) {
      v16bf bv;
      int n = nt * 16 + (lane & 15);
#pragma unroll
      for (int h = 0; h < 16; ++h) {
        int key = c0 + ((lane >> 4) << 4) + h;
        bv[h] = Vrow[(size_t)key * HD + n];
      }
      cacc[nt] = wmma_bf16(pa, bv, cacc[nt]);
    }
  }
#pragma unroll
  for (int nt = 0; nt < 4; ++nt)
#pragma unroll
    for (int r = 0; r < 8; ++r) {
      int row = q0 + r + ((lane >> 4) << 3);     // seq pos s
      int d = nt * 16 + (lane & 15);
      size_t t = (size_t)row * BSZ + bidx;       // token = s*B + b
      ctxb[t * EMB + head * HD + d] = f2bf(cacc[nt][r]);
    }
}

// ---------------------------------------------------------------------------
// K5: weights[b] = sum_heads softmax(...) / 16, NO atomics.
// Block = (q-tile 128 rows, 64-col key range, batch); head-sum accumulated
// in a 4x8 register tile; one coalesced store per element.
// ---------------------------------------------------------------------------
__global__ __launch_bounds__(256) void weights_kernel(
    const __bf16* __restrict__ Qb, const __bf16* __restrict__ Kb,
    const float* __restrict__ mask, const float* __restrict__ stats,
    float* __restrict__ weights) {
  const int tid = threadIdx.x, lane = tid & 31, wv = tid >> 5;
  const int b = blockIdx.z;
  const int q0 = blockIdx.x * 128 + wv * 16;
  const int kr0 = blockIdx.y * 64;
  const int colbase = kr0 + (lane & 15);

  float mtile[4][8];
#pragma unroll
  for (int kt = 0; kt < 4; ++kt)
#pragma unroll
    for (int r = 0; r < 8; ++r) {
      int row = q0 + r + ((lane >> 4) << 3);
      mtile[kt][r] = mask[(size_t)row * S_TOT + colbase + kt * 16];
    }

  float wacc[4][8] = {};
  for (int head = 0; head < HEADS; ++head) {
    int bh = b * HEADS + head;
    const __bf16* Qrow = Qb + (size_t)bh * S_TOT * HD;
    const __bf16* Krow = Kb + (size_t)bh * S_TOT * HD;
    v16bf a0, a1;
    int qr = q0 + (lane & 15);
#pragma unroll
    for (int h = 0; h < 16; ++h) {
      int k = a_kidx(h, lane);
      a0[h] = Qrow[(size_t)qr * HD + k];
      a1[h] = Qrow[(size_t)qr * HD + 32 + k];
    }
    float M[8], IL[8];
#pragma unroll
    for (int r = 0; r < 8; ++r) {
      int row = q0 + r + ((lane >> 4) << 3);
      M[r] = stats[((size_t)bh * S_TOT + row) * 2 + 0];
      float l = stats[((size_t)bh * S_TOT + row) * 2 + 1];
      IL[r] = (l > 0.f) ? (0.0625f / l) : 0.f;   // fold /HEADS into 1/L
    }
#pragma unroll
    for (int kt = 0; kt < 4; ++kt) {
      int key = kr0 + kt * 16 + (lane & 15);
      const __bf16* kp = &Krow[(size_t)key * HD + ((lane >> 4) << 4)];
      v16bf b0 = *reinterpret_cast<const v16bf*>(kp);
      v16bf b1 = *reinterpret_cast<const v16bf*>(kp + 32);
      v8f z = {};
      v8f sc = wmma_bf16(a0, b0, z);
      sc = wmma_bf16(a1, b1, sc);
#pragma unroll
      for (int r = 0; r < 8; ++r)
        wacc[kt][r] += __expf(sc[r] + mtile[kt][r] - M[r]) * IL[r];
    }
  }
#pragma unroll
  for (int kt = 0; kt < 4; ++kt)
#pragma unroll
    for (int r = 0; r < 8; ++r) {
      int row = q0 + r + ((lane >> 4) << 3);
      weights[((size_t)b * S_TOT + row) * S_TOT + colbase + kt * 16] =
          wacc[kt][r];
    }
}

// ---------------------------------------------------------------------------
// K4: out-projection per segment: ctx (tokens x E) @ W^T + b -> fp32 out.
// ---------------------------------------------------------------------------
__global__ __launch_bounds__(256) void outproj_kernel(
    const __bf16* __restrict__ ctxb, const float* __restrict__ w,
    const float* __restrict__ bias, float* __restrict__ out, int tok_base) {
  __shared__ __align__(32) __bf16 Alds[128 * 32];
  __shared__ __align__(32) __bf16 Blds[64 * 32];
  const int tid = threadIdx.x, lane = tid & 31, wv = tid >> 5;
  const int m0 = blockIdx.x * 128;
  const int n0 = blockIdx.y * 64;
  const __bf16* X = ctxb + (size_t)tok_base * EMB;

  v8f acc[4] = {};
  for (int k0 = 0; k0 < EMB; k0 += 32) {
    for (int idx = tid; idx < 128 * 32; idx += 256) {
      int i = idx >> 5, j = idx & 31;
      Alds[idx] = X[(size_t)(m0 + i) * EMB + k0 + j];
    }
    for (int idx = tid; idx < 64 * 32; idx += 256) {
      int n = idx >> 5, k = idx & 31;
      Blds[idx] = f2bf(w[(size_t)(n0 + n) * EMB + k0 + k]);
    }
    int kn = (k0 + 32 < EMB) ? k0 + 32 : k0;
    __builtin_prefetch(&X[(size_t)(m0 + (tid >> 1)) * EMB + kn + (tid & 1) * 16], 0, 3);
    __builtin_prefetch(&w[(size_t)(n0 + (tid >> 2)) * EMB + kn + (tid & 3) * 8], 0, 3);
    __syncthreads();
    v16bf a;
#pragma unroll
    for (int h = 0; h < 16; ++h)
      a[h] = Alds[(wv * 16 + (lane & 15)) * 32 + a_kidx(h, lane)];
    v16bf bfr[4];
#pragma unroll
    for (int nt = 0; nt < 4; ++nt)
      bfr[nt] = *reinterpret_cast<const v16bf*>(
          &Blds[(nt * 16 + (lane & 15)) * 32 + ((lane >> 4) << 4)]);
#pragma unroll
    for (int nt = 0; nt < 4; ++nt)
      acc[nt] = wmma_bf16(a, bfr[nt], acc[nt]);
    __syncthreads();
  }
#pragma unroll
  for (int nt = 0; nt < 4; ++nt)
#pragma unroll
    for (int r = 0; r < 8; ++r) {
      int m = m0 + wv * 16 + r + ((lane >> 4) << 3);
      int n = n0 + nt * 16 + (lane & 15);
      out[(size_t)(tok_base + m) * EMB + n] = acc[nt][r] + bias[n];
    }
}

extern "C" void kernel_launch(void* const* d_in, const int* in_sizes, int n_in,
                              void* d_out, int out_size, void* d_ws,
                              size_t ws_size, hipStream_t stream) {
  const float* query = (const float*)d_in[0];
  const float* key   = (const float*)d_in[1];
  const float* value = (const float*)d_in[2];
  const float* mask  = (const float*)d_in[3];
  const float* in_w[3]  = {(const float*)d_in[4],  (const float*)d_in[8],  (const float*)d_in[12]};
  const float* in_b[3]  = {(const float*)d_in[5],  (const float*)d_in[9],  (const float*)d_in[13]};
  const float* out_w[3] = {(const float*)d_in[6],  (const float*)d_in[10], (const float*)d_in[14]};
  const float* out_b[3] = {(const float*)d_in[7],  (const float*)d_in[11], (const float*)d_in[15]};

  char* ws = (char*)d_ws;                 // needs 130 MiB
  __bf16* Qb   = (__bf16*)(ws + ((size_t)0  << 20));
  __bf16* Kb   = (__bf16*)(ws + ((size_t)32 << 20));
  __bf16* Vb   = (__bf16*)(ws + ((size_t)64 << 20));
  __bf16* ctxb = (__bf16*)(ws + ((size_t)96 << 20));
  float*  stats = (float*)(ws + ((size_t)128 << 20));

  float* attn_out = (float*)d_out;                              // S*B*E
  float* weights  = attn_out + (size_t)S_TOT * BSZ * EMB;       // B*S*S

  const int offs[4] = {0, 256, 512, 1024};
  for (int sg = 0; sg < 3; ++sg) {
    int L = offs[sg + 1] - offs[sg];
    dim3 g(L * BSZ / 128, 48);
    proj_kernel<<<g, 256, 0, stream>>>(query, key, value, in_w[sg], in_b[sg],
                                       Qb, Kb, Vb, offs[sg] * BSZ);
  }
  stats_kernel<<<dim3(8, 256), 256, 0, stream>>>(Qb, Kb, mask, stats);
  attn_kernel<<<dim3(8, 256), 256, 0, stream>>>(Qb, Kb, Vb, mask, stats, ctxb);
  weights_kernel<<<dim3(8, 16, 16), 256, 0, stream>>>(Qb, Kb, mask, stats,
                                                      weights);
  for (int sg = 0; sg < 3; ++sg) {
    int L = offs[sg + 1] - offs[sg];
    dim3 g(L * BSZ / 128, 16);
    outproj_kernel<<<g, 256, 0, stream>>>(ctxb, out_w[sg], out_b[sg], attn_out,
                                          offs[sg] * BSZ);
  }
}
